// WanSelfAttention_45646912421898
// MI455X (gfx1250) — matrix operand order repaired and checked by reading, not compiled, not run on hardware
//
#include <hip/hip_runtime.h>
#include <hip/hip_bf16.h>

#define DIM 1536
#define NH 12
#define HD 128
#define L_TOK 6400
#define LK 3200

typedef __bf16 bf16_t;
typedef __attribute__((ext_vector_type(16))) __bf16 v16bf;
typedef __attribute__((ext_vector_type(8))) float v8f;

union BF16x16 {
  v16bf v;
  __bf16 e[16];
  uint4 q[2];
};

__device__ __forceinline__ v8f wmma_bf16(v16bf a, v16bf b, v8f c) {
  // D = A(16x32) * B(32x16) + C, f32 accumulate
  return __builtin_amdgcn_wmma_f32_16x16x32_bf16(false, a, false, b, (short)0, c,
                                                 false, false);
}

__device__ __forceinline__ uint4 ld128(const bf16_t* p) {
  return *(const uint4*)p;
}

// Async copy 16 bytes global -> LDS (CDNA5 async path, tracked by ASYNCcnt).
// Low 32 bits of a generic pointer to __shared__ are the LDS byte offset
// (see ISA 10.2).
__device__ __forceinline__ void async_g2l_b128(const bf16_t* gsrc, bf16_t* ldst) {
  unsigned lds_off = (unsigned)(size_t)ldst;
  unsigned long long gaddr = (unsigned long long)(size_t)gsrc;
  asm volatile("global_load_async_to_lds_b128 %0, %1, off"
               :: "v"(lds_off), "v"(gaddr)
               : "memory");
}

__device__ __forceinline__ void wait_async_le1() {
  asm volatile("s_wait_asynccnt 0x1" ::: "memory");
}
__device__ __forceinline__ void wait_async_le0() {
  asm volatile("s_wait_asynccnt 0x0" ::: "memory");
}

// Raw workgroup split-barrier: no implicit loadcnt/storecnt drain (we manage
// LDS producer/consumer ordering explicitly with s_wait_asynccnt and the
// data dependences of the DS reads). Keeps next-step global loads in flight.
__device__ __forceinline__ void wg_barrier() {
  asm volatile("s_barrier_signal -1\n\ts_barrier_wait -1" ::: "memory");
}

// ---------------------------------------------------------------------------
// Elementwise conversions
// ---------------------------------------------------------------------------
__global__ __launch_bounds__(256) void cvt_bf16(const float* __restrict__ in,
                                                bf16_t* __restrict__ out, int n) {
  int i = blockIdx.x * 256 + threadIdx.x;
  if (i < n) out[i] = (bf16_t)in[i];
}

// Wt[n][k] = (bf16) W[k][n]   (DIM x DIM)
__global__ __launch_bounds__(256) void transpose_w(const float* __restrict__ W,
                                                   bf16_t* __restrict__ Wt) {
  int idx = blockIdx.x * 256 + threadIdx.x;
  if (idx >= DIM * DIM) return;
  int n = idx / DIM, k = idx % DIM;
  Wt[idx] = (bf16_t)W[(size_t)k * DIM + n];
}

// ---------------------------------------------------------------------------
// GEMM: C[M][N] = A_bf16[M][K] * Wt_bf16[N][K]^T + bias[N]   (f32 out)
// Block (256 thr / 8 waves) computes a 128x64 tile. The 64-col x 32-k B strip
// is double-buffered in LDS via async global->LDS copies (shared by the 8
// waves); the k-loop is software-pipelined by 2 so the steady state is
// branch-free with no register rotation and no loadcnt drain at barriers.
// ---------------------------------------------------------------------------
__device__ __forceinline__ void gemm_step(const bf16_t* bbase, const BF16x16& a,
                                          v8f acc[4], int col, int hi) {
#pragma unroll
  for (int t = 0; t < 4; ++t) {
    const bf16_t* pb = bbase + (t * 16 + col) * 32 + (hi ? 8 : 0);
    BF16x16 b;
    b.q[0] = *(const uint4*)pb;        // ds_load_b128
    b.q[1] = *(const uint4*)(pb + 16);
    acc[t] = wmma_bf16(a.v, b.v, acc[t]);
  }
}

__global__ __launch_bounds__(256) void gemm_bf16_lds(const bf16_t* __restrict__ A,
                                                     const bf16_t* __restrict__ Bt,
                                                     const float* __restrict__ bias,
                                                     float* __restrict__ C,
                                                     int M, int N, int K) {
  __shared__ alignas(16) bf16_t Bs[2][64 * 32];  // [buf][n_local*32 + k]  (8 KB)

  const int tn = N / 64;
  const int tile_n = blockIdx.x % tn;
  const int m_base = (blockIdx.x / tn) * 128;
  const int tid = threadIdx.x;
  const int wave = tid >> 5;
  const int lane = tid & 31;
  const int col = lane & 15;
  const int hi = lane >> 4;
  const int n0 = tile_n * 64;

  // B staging map: 256 threads x 16B = 64 cols x 32 k of bf16
  const int bn = tid >> 2;        // 0..63 : local column n
  const int bk = (tid & 3) * 8;   // 0,8,16,24 : k sub-chunk
  const bf16_t* bsrc = Bt + (size_t)(n0 + bn) * K + bk;
  bf16_t* bdst0 = &Bs[0][bn * 32 + bk];
  bf16_t* bdst1 = &Bs[1][bn * 32 + bk];

  const bf16_t* pa0 = A + (size_t)(m_base + wave * 16 + col) * K;
  const int alo = hi ? 8 : 0;
  const int ahi = hi ? 24 : 16;

  v8f zero = {0.f, 0.f, 0.f, 0.f, 0.f, 0.f, 0.f, 0.f};
  v8f acc[4];
#pragma unroll
  for (int t = 0; t < 4; ++t) acc[t] = zero;

  // prologue: stage both buffers, load both A tiles
  async_g2l_b128(bsrc, bdst0);
  BF16x16 aE, aO;
  aE.q[0] = ld128(pa0 + alo);
  aE.q[1] = ld128(pa0 + ahi);
  async_g2l_b128(bsrc + 32, bdst1);
  aO.q[0] = ld128(pa0 + 32 + alo);
  aO.q[1] = ld128(pa0 + 32 + ahi);

  int k0 = 0;
  for (; k0 + 64 < K; k0 += 64) {
    // ---- even step: k0, buf0 ----
    wait_async_le1();      // buf0 complete (buf1 still in flight)
    wg_barrier();
    gemm_step(&Bs[0][0], aE, acc, col, hi);
    aE.q[0] = ld128(pa0 + k0 + 64 + alo);   // A for next even step (stays in flight)
    aE.q[1] = ld128(pa0 + k0 + 64 + ahi);
    wg_barrier();          // buf0 fully consumed
    async_g2l_b128(bsrc + k0 + 64, bdst0);
    // ---- odd step: k0+32, buf1 ----
    wait_async_le1();      // buf1 complete (new buf0 copy in flight)
    wg_barrier();
    gemm_step(&Bs[1][0], aO, acc, col, hi);
    aO.q[0] = ld128(pa0 + k0 + 96 + alo);   // A for next odd step
    aO.q[1] = ld128(pa0 + k0 + 96 + ahi);
    wg_barrier();          // buf1 fully consumed
    async_g2l_b128(bsrc + k0 + 96, bdst1);
  }
  // tail: k0 == K-64, both buffers staged, no refills
  wait_async_le1();
  wg_barrier();
  gemm_step(&Bs[0][0], aE, acc, col, hi);
  wait_async_le0();
  wg_barrier();
  gemm_step(&Bs[1][0], aO, acc, col, hi);

#pragma unroll
  for (int t = 0; t < 4; ++t) {
    int n = n0 + t * 16 + col;
    float bsv = bias[n];
#pragma unroll
    for (int j = 0; j < 8; ++j) {
      int m = m_base + wave * 16 + j + (hi ? 8 : 0);
      C[(size_t)m * N + n] = acc[t][j] + bsv;
    }
  }
}

// ---------------------------------------------------------------------------
// RMSNorm (over DIM) + per-head RoPE; writes head-major bf16.
// subsample==1: keep only even-frame tokens (k path), Lout = 3200.
// ---------------------------------------------------------------------------
__global__ __launch_bounds__(256) void rmsnorm_rope(const float* __restrict__ Y,
                                                    const float* __restrict__ g,
                                                    const float* __restrict__ freqs,
                                                    bf16_t* __restrict__ out,
                                                    int subsample) {
  int token = blockIdx.x;
  if (subsample && ((token >> 6) & 1)) return;  // odd frame: dropped
  int tid = threadIdx.x;
  __shared__ float red[256];
  const float* yrow = Y + (size_t)token * DIM;
  float s = 0.f;
  for (int i = tid; i < DIM; i += 256) {
    float v = yrow[i];
    s += v * v;
  }
  red[tid] = s;
  __syncthreads();
  for (int off = 128; off > 0; off >>= 1) {
    if (tid < off) red[tid] += red[tid + off];
    __syncthreads();
  }
  float rms = rsqrtf(red[0] * (1.0f / DIM) + 1e-6f);

  int fi = token >> 6, hrow = (token >> 3) & 7, wrow = token & 7;
  int tok_out = subsample ? ((token >> 7) * 64 + (token & 63)) : token;
  int Lout = subsample ? LK : L_TOK;

  for (int p = tid; p < NH * 64; p += 256) {
    int head = p >> 6, i = p & 63;
    int r = (i < 22) ? fi : ((i < 43) ? hrow : wrow);
    float ang = freqs[r * 64 + i];
    float cs = __cosf(ang), sn = __sinf(ang);
    int base = head * HD + 2 * i;
    float xr = yrow[base] * rms * g[base];
    float xi = yrow[base + 1] * rms * g[base + 1];
    size_t ob = (size_t)head * Lout * HD + (size_t)tok_out * HD + 2 * i;
    out[ob] = (bf16_t)(xr * cs - xi * sn);
    out[ob + 1] = (bf16_t)(xr * sn + xi * cs);
  }
}

// v_t[head][d][kt] = (bf16) Yv[token(kt)][head*128+d], even frames only
__global__ __launch_bounds__(256) void v_cast(const float* __restrict__ Y,
                                              bf16_t* __restrict__ vt) {
  int idx = blockIdx.x * 256 + threadIdx.x;
  if (idx >= NH * HD * LK) return;
  int head = idx / (HD * LK);
  int r = idx % (HD * LK);
  int d = r / LK;
  int kt = r % LK;
  int token = (kt >> 6) * 128 + (kt & 63);
  vt[idx] = (bf16_t)Y[(size_t)token * DIM + head * HD + d];
}

// ---------------------------------------------------------------------------
// Flash attention: one wave per (head, 16-query tile).
// S' = K_tile(16x d) x Q^T(d x16) so S' C-layout == A-layout for P@V.
// V tiles are preloaded before the softmax VALU block so their latency
// overlaps the exp2/reduction math and the P@V WMMAs run back-to-back.
// ---------------------------------------------------------------------------
__global__ __launch_bounds__(32) void attn_kernel(const bf16_t* __restrict__ qg,
                                                  const bf16_t* __restrict__ kg,
                                                  const bf16_t* __restrict__ vg,
                                                  bf16_t* __restrict__ og) {
  const int head = blockIdx.x / (L_TOK / 16);
  const int q0 = (blockIdx.x % (L_TOK / 16)) * 16;
  const int lane = threadIdx.x;
  const int col = lane & 15;
  const int hi = lane >> 4;

  const bf16_t* qp = qg + (size_t)head * L_TOK * HD;
  const bf16_t* kp = kg + (size_t)head * LK * HD;
  const bf16_t* vp = vg + (size_t)head * HD * LK;

  // Q as B operand (Q^T), split over 4 K=32 chunks of d
  BF16x16 qB[4];
#pragma unroll
  for (int c = 0; c < 4; ++c) {
    const bf16_t* p = qp + (size_t)(q0 + col) * HD + 32 * c + (hi ? 8 : 0);
    qB[c].q[0] = ld128(p);
    qB[c].q[1] = ld128(p + 16);
  }

  const float sc2 = 0.1275174245f;  // log2(e) / sqrt(128)
  float m_run = -INFINITY;
  float l_run = 0.f;
  v8f zero = {0.f, 0.f, 0.f, 0.f, 0.f, 0.f, 0.f, 0.f};
  v8f O[8];
#pragma unroll
  for (int c = 0; c < 8; ++c) O[c] = zero;

  for (int kt = 0; kt < LK; kt += 32) {
    v8f S0 = zero, S1 = zero;  // rows=keys, cols=q
#pragma unroll
    for (int c = 0; c < 4; ++c) {
      BF16x16 a0, a1;
      const bf16_t* pk = kp + (size_t)(kt + col) * HD + 32 * c;
      a0.q[0] = ld128(pk + (hi ? 8 : 0));
      a0.q[1] = ld128(pk + (hi ? 24 : 16));
      const bf16_t* pk1 = pk + 16 * HD;
      a1.q[0] = ld128(pk1 + (hi ? 8 : 0));
      a1.q[1] = ld128(pk1 + (hi ? 24 : 16));
      S0 = wmma_bf16(a0.v, qB[c].v, S0);
      S1 = wmma_bf16(a1.v, qB[c].v, S1);
    }

    // Preload this tile's V B-operands; latency hidden under softmax math.
    BF16x16 vB[8];
#pragma unroll
    for (int c = 0; c < 8; ++c) {
      const bf16_t* pv = vp + (size_t)(16 * c + col) * LK + kt + (hi ? 8 : 0);
      vB[c].q[0] = ld128(pv);
      vB[c].q[1] = ld128(pv + 16);
    }

    // online softmax stats; this lane covers q = col, 16 of 32 keys
    float mloc = S0[0];
#pragma unroll
    for (int j = 0; j < 8; ++j) {
      mloc = fmaxf(mloc, S0[j]);
      mloc = fmaxf(mloc, S1[j]);
    }
    mloc = fmaxf(mloc, __shfl_xor(mloc, 16, 32));
    float m_new = fmaxf(m_run, mloc * sc2);
    float cexp = exp2f(m_run - m_new);

    BF16x16 P;  // already in A-matrix layout (rows=q, K=key)
    float lsum = 0.f;
#pragma unroll
    for (int j = 0; j < 8; ++j) {
      float p0 = exp2f(S0[j] * sc2 - m_new);
      float p1 = exp2f(S1[j] * sc2 - m_new);
      P.e[j] = (bf16_t)p0;
      P.e[8 + j] = (bf16_t)p1;
      lsum += p0 + p1;
    }
    lsum += __shfl_xor(lsum, 16, 32);
    l_run = l_run * cexp + lsum;
    m_run = m_new;

    // rescale O rows by cexp (row q = j + (hi?8:0), stats live in lane q)
#pragma unroll
    for (int j = 0; j < 8; ++j) {
      float cj = __shfl(cexp, (hi ? 8 : 0) + j, 32);
#pragma unroll
      for (int c = 0; c < 8; ++c) O[c][j] *= cj;
    }
    // O += P(16x32) * V(32x128) over 8 column chunks, operands already resident
#pragma unroll
    for (int c = 0; c < 8; ++c) {
      O[c] = wmma_bf16(P.v, vB[c].v, O[c]);
    }
  }
  float inv = 1.f / l_run;
#pragma unroll
  for (int j = 0; j < 8; ++j) {
    float ij = __shfl(inv, (hi ? 8 : 0) + j, 32);
    int row = q0 + j + (hi ? 8 : 0);
#pragma unroll
    for (int c = 0; c < 8; ++c) {
      og[(size_t)row * DIM + head * HD + 16 * c + col] = (bf16_t)(O[c][j] * ij);
    }
  }
}

// ---------------------------------------------------------------------------
extern "C" void kernel_launch(void* const* d_in, const int* in_sizes, int n_in,
                              void* d_out, int out_size, void* d_ws, size_t ws_size,
                              hipStream_t stream) {
  (void)in_sizes; (void)n_in; (void)out_size; (void)ws_size;
  const float* x     = (const float*)d_in[0];
  const float* freqs = (const float*)d_in[1];
  const float* Wq = (const float*)d_in[2];  const float* bq = (const float*)d_in[3];
  const float* Wk = (const float*)d_in[4];  const float* bk = (const float*)d_in[5];
  const float* Wv = (const float*)d_in[6];  const float* bv = (const float*)d_in[7];
  const float* Wo = (const float*)d_in[8];  const float* bo = (const float*)d_in[9];
  const float* gq = (const float*)d_in[10]; const float* gk = (const float*)d_in[11];
  float* out = (float*)d_out;

  char* ws = (char*)d_ws;
  size_t off = 0;
  bf16_t* xb = (bf16_t*)(ws + off); off += (size_t)L_TOK * DIM * 2;   // 19.7 MB
  bf16_t* wt = (bf16_t*)(ws + off); off += (size_t)DIM * DIM * 2;     //  4.7 MB
  float*  Y  = (float*)(ws + off);  off += (size_t)L_TOK * DIM * 4;   // 39.3 MB
  bf16_t* qb = (bf16_t*)(ws + off); off += (size_t)NH * L_TOK * HD * 2;
  bf16_t* kb = (bf16_t*)(ws + off); off += (size_t)NH * LK * HD * 2;
  bf16_t* vt = (bf16_t*)(ws + off); off += (size_t)NH * HD * LK * 2;
  bf16_t* attnb = (bf16_t*)Y;  // Y no longer needed once v_cast is done

  const int gemm_blocks = (L_TOK / 128) * (DIM / 64);  // 50 * 24 = 1200
  const int tw_blocks = (DIM * DIM + 255) / 256;

  cvt_bf16<<<(L_TOK * DIM + 255) / 256, 256, 0, stream>>>(x, xb, L_TOK * DIM);

  // Q projection -> RMSNorm + RoPE
  transpose_w<<<tw_blocks, 256, 0, stream>>>(Wq, wt);
  gemm_bf16_lds<<<gemm_blocks, 256, 0, stream>>>(xb, wt, bq, Y, L_TOK, DIM, DIM);
  rmsnorm_rope<<<L_TOK, 256, 0, stream>>>(Y, gq, freqs, qb, 0);

  // K projection -> RMSNorm + RoPE + frame subsample
  transpose_w<<<tw_blocks, 256, 0, stream>>>(Wk, wt);
  gemm_bf16_lds<<<gemm_blocks, 256, 0, stream>>>(xb, wt, bk, Y, L_TOK, DIM, DIM);
  rmsnorm_rope<<<L_TOK, 256, 0, stream>>>(Y, gk, freqs, kb, 1);

  // V projection -> transposed subsampled bf16
  transpose_w<<<tw_blocks, 256, 0, stream>>>(Wv, wt);
  gemm_bf16_lds<<<gemm_blocks, 256, 0, stream>>>(xb, wt, bv, Y, L_TOK, DIM, DIM);
  v_cast<<<(NH * HD * LK + 255) / 256, 256, 0, stream>>>(Y, vt);

  // Attention
  attn_kernel<<<NH * (L_TOK / 16), 32, 0, stream>>>(qb, kb, vt, attnb);

  // Output projection (f32 result with bias) -> d_out
  transpose_w<<<tw_blocks, 256, 0, stream>>>(Wo, wt);
  gemm_bf16_lds<<<gemm_blocks, 256, 0, stream>>>(attnb, wt, bo, out, L_TOK, DIM, DIM);
}
